// Block_3298534883752
// MI455X (gfx1250) — matrix operand order repaired and checked by reading, not compiled, not run on hardware
//
#include <hip/hip_runtime.h>
#include <hip/hip_bf16.h>
#include <math.h>

// ---------------------------------------------------------------------------
// CDNA5 (gfx1250) wave32 WMMA types & helpers
// ---------------------------------------------------------------------------
typedef __attribute__((ext_vector_type(16))) __bf16 v16bf;
typedef __attribute__((ext_vector_type(8)))  float  v8f;
typedef __attribute__((ext_vector_type(4)))  unsigned int u32x4;   // 16B chunk
typedef __attribute__((ext_vector_type(8)))  int i32x8;
typedef __attribute__((ext_vector_type(4)))  int i32x4;
typedef __attribute__((address_space(3)))    void lds_void;

union Frag { v16bf v; u32x4 q[2]; };

__device__ __forceinline__ v8f wmma_bf16(const Frag& a, const Frag& b, v8f c) {
  // D = A(16x32 bf16) * B(32x16 bf16) + C(16x16 f32)
  return __builtin_amdgcn_wmma_f32_16x16x32_bf16(
      /*neg_a=*/false, a.v, /*neg_b=*/false, b.v,
      /*c_mod=*/(short)0, c, /*reuse_a=*/false, /*reuse_b=*/false);
}

// A fragment (16x32, 16-bit): lane L -> row m = L%16, half = L/16.
//   elements 0..7  hold K = 8*half + 0..7      (one b128)
//   elements 8..15 hold K = 16 + 8*half + 0..7 (one b128)
// Source layout: row-major [m][k], stride `ld` elements.
__device__ __forceinline__ void load_frag_a(Frag& f, const __bf16* base,
                                            int m, int hlf, int ld) {
  const __bf16* p = base + (size_t)m * ld + 8 * hlf;
  f.q[0] = *(const u32x4*)(p);
  f.q[1] = *(const u32x4*)(p + 16);
}

// B fragment (32x16, 16-bit): lane L -> col n = L%16, half = L/16.
//   elements 0..7  hold K = 8*half + 0..7, elements 8..15 hold K = 16 + 8*half + 0..7
// Source layout: n-major [n][k], stride `ld` elements.
__device__ __forceinline__ void load_frag_b(Frag& f, const __bf16* base,
                                            int n, int hlf, int ld) {
  const __bf16* p = base + (size_t)n * ld + 8 * hlf;
  f.q[0] = *(const u32x4*)(p);
  f.q[1] = *(const u32x4*)(p + 16);
}

// LDS byte offset of a generic pointer into shared memory (AS3 ptrs are 32-bit
// group-segment offsets on amdgcn).
__device__ __forceinline__ unsigned lds_offset(void* p) {
  return (unsigned)(size_t)(lds_void*)p;
}

// ---------------------------------------------------------------------------
// DPP16 width-16 XOR butterfly reductions (pure VALU, no LDS traffic).
// xor15=ROW_MIRROR(0x140), xor7=ROW_HALF_MIRROR(0x141),
// xor3=quad_perm(3,2,1,0)=0x1B, xor1=quad_perm(1,0,3,2)=0xB1.
// {15,7,3,1} generates the full 16-lane group; all ctrls stay within a DPP
// row, so lanes 0-15 and 16-31 reduce independently (matches C-frag halves).
// ---------------------------------------------------------------------------
__device__ __forceinline__ float dpp_perm_f(float x, const int ctrl) {
  int xi = __builtin_bit_cast(int, x);
  int r;
  switch (ctrl) {  // ctrl must be an ICE for the builtin
    case 0x140: r = __builtin_amdgcn_update_dpp(xi, xi, 0x140, 0xF, 0xF, true); break;
    case 0x141: r = __builtin_amdgcn_update_dpp(xi, xi, 0x141, 0xF, 0xF, true); break;
    case 0x01B: r = __builtin_amdgcn_update_dpp(xi, xi, 0x01B, 0xF, 0xF, true); break;
    default:    r = __builtin_amdgcn_update_dpp(xi, xi, 0x0B1, 0xF, 0xF, true); break;
  }
  return __builtin_bit_cast(float, r);
}
__device__ __forceinline__ float row16_max(float v) {
  v = fmaxf(v, dpp_perm_f(v, 0x140));
  v = fmaxf(v, dpp_perm_f(v, 0x141));
  v = fmaxf(v, dpp_perm_f(v, 0x01B));
  v = fmaxf(v, dpp_perm_f(v, 0x0B1));
  return v;
}
__device__ __forceinline__ float row16_sum(float v) {
  v += dpp_perm_f(v, 0x140);
  v += dpp_perm_f(v, 0x141);
  v += dpp_perm_f(v, 0x01B);
  v += dpp_perm_f(v, 0x0B1);
  return v;
}

// ---------------------------------------------------------------------------
// Tensor Data Mover: async 2-D tile load Global -> LDS.
// D# packing per CDNA5 ISA (§8.3 group0, §8.4 group1).
// ---------------------------------------------------------------------------
__device__ __forceinline__ void tdm_load_2d_bf16(
    const __bf16* gptr, unsigned lds_byte_off,
    unsigned tile_d0, unsigned tile_d1,
    unsigned tensor_d0, unsigned tensor_d1, unsigned stride_d0,
    unsigned pad_en, unsigned pad_interval, unsigned pad_amount) {
  unsigned long long ga = (unsigned long long)(size_t)gptr;
  u32x4 g0;
  g0[0] = 1u;                                       // count=1, user mode
  g0[1] = lds_byte_off;                             // lds_addr [63:32]
  g0[2] = (unsigned)(ga & 0xFFFFFFFFu);             // global_addr lo
  g0[3] = (unsigned)((ga >> 32) & 0x01FFFFFFu)      // global_addr hi (57b)
          | (2u << 30);                             // type=2 ("image")
  i32x8 g1;
  g1[0] = (int)((1u << 16)                          // data_size=1 (2 bytes)
                | (pad_en << 20)
                | (pad_interval << 22)
                | (pad_amount << 25));
  g1[1] = (int)((tensor_d0 & 0xFFFFu) << 16);       // tensor_dim0 [79:48]
  g1[2] = (int)(((tensor_d0 >> 16) & 0xFFFFu)
                | ((tensor_d1 & 0xFFFFu) << 16));   // tensor_dim1 [111:80]
  g1[3] = (int)(((tensor_d1 >> 16) & 0xFFFFu)
                | (tile_d0 << 16));                 // tile_dim0 [127:112]
  g1[4] = (int)tile_d1;                             // tile_dim1; tile_dim2=0
  g1[5] = (int)stride_d0;                           // tensor_dim0_stride lo
  g1[6] = 0;
  g1[7] = 0;
  i32x4 z4 = {0, 0, 0, 0};
#if __clang_major__ >= 23
  i32x8 z8 = {0, 0, 0, 0, 0, 0, 0, 0};
  __builtin_amdgcn_tensor_load_to_lds(g0, g1, z4, z4, z8, 0);
#else
  __builtin_amdgcn_tensor_load_to_lds(g0, g1, z4, z4, 0);
#endif
}

// ---------------------------------------------------------------------------
// f32 -> bf16 conversion (weights)
// ---------------------------------------------------------------------------
__global__ void cvt_bf16_kernel(const float* __restrict__ in,
                                __bf16* __restrict__ out, int n) {
  int i = blockIdx.x * 256 + threadIdx.x;
  if (i < n) out[i] = (__bf16)in[i];
}

// ---------------------------------------------------------------------------
// Row LayerNorm (block per row) -> bf16
// ---------------------------------------------------------------------------
__global__ __launch_bounds__(256)
void ln_kernel(const float* __restrict__ x, const float* __restrict__ w,
               const float* __restrict__ b, __bf16* __restrict__ out, int C) {
  const int row = blockIdx.x;
  const float* xr = x + (size_t)row * C;
  __shared__ float red[256];
  float s = 0.f;
  for (int i = threadIdx.x; i < C; i += 256) s += xr[i];
  red[threadIdx.x] = s; __syncthreads();
  for (int st = 128; st > 0; st >>= 1) {
    if (threadIdx.x < st) red[threadIdx.x] += red[threadIdx.x + st];
    __syncthreads();
  }
  const float mu = red[0] / (float)C;
  __syncthreads();
  float v = 0.f;
  for (int i = threadIdx.x; i < C; i += 256) { float d = xr[i] - mu; v += d * d; }
  red[threadIdx.x] = v; __syncthreads();
  for (int st = 128; st > 0; st >>= 1) {
    if (threadIdx.x < st) red[threadIdx.x] += red[threadIdx.x + st];
    __syncthreads();
  }
  const float rstd = rsqrtf(red[0] / (float)C + 1e-5f);
  __syncthreads();
  for (int i = threadIdx.x; i < C; i += 256)
    out[(size_t)row * C + i] = (__bf16)((xr[i] - mu) * rstd * w[i] + b[i]);
}

// ---------------------------------------------------------------------------
// Generic bf16 WMMA GEMM: out = epilogue(oscale * (A[M,K] @ B[K,N]))
//   block tile 128x128, 8 waves (2x4), wave tile 64x32, K-step 32.
//   Software-pipelined double buffer: TDM for tile i+1 issued before the
//   WMMA burst of tile i (s_wait_tensorcnt only afterwards); B-tile global
//   loads for i+1 land in registers before compute, LDS scatter after.
//   One barrier per k-step.
// ---------------------------------------------------------------------------
#define G_LDA 40   // 32 + 8 pad; 40*2 B stride keeps b128 frag loads aligned
#define G_LDB 40

template <bool BIAS, bool RES, bool GELU, bool OUTF>
__global__ __launch_bounds__(256)
void gemm_bf16_kernel(const __bf16* __restrict__ A, const __bf16* __restrict__ B,
                      int M, int N, int K, float oscale,
                      const float* __restrict__ bias,
                      const float* __restrict__ res,
                      float* __restrict__ outF, __bf16* __restrict__ outB) {
  __shared__ __attribute__((aligned(16))) __bf16 sA[2][128 * G_LDA]; // [m][k]
  __shared__ __attribute__((aligned(16))) __bf16 sB[2][128 * G_LDB]; // [n][k]

  const int tid = threadIdx.x;
  const int wid = tid >> 5, lane = tid & 31;
  const int hlf = lane >> 4, ln = lane & 15;
  const int wm = wid >> 2, wn = wid & 3;            // 2 x 4 wave grid
  const int m0 = blockIdx.x * 128, n0 = blockIdx.y * 128;
  const unsigned sA_off[2] = {lds_offset(sA[0]), lds_offset(sA[1])};

  // B-tile copy indices: 2 chunks of 8 bf16 per thread
  const int br0 = tid >> 4,          bc0 = (tid & 15) * 8;
  const int br1 = (tid + 256) >> 4,  bc1 = bc0;

  v8f acc[4][2] = {};
  u32x4 breg[2];

  const int T = K / 32;

  // ---- prologue: tile 0 ----
  if (wid == 0)
    tdm_load_2d_bf16(A + (size_t)m0 * K, sA_off[0],
                     32, 128, (unsigned)K, (unsigned)M, (unsigned)K,
                     1, 3, 3);                      // pad 4 DW per 16 DW
  breg[0] = *(const u32x4*)(B + (size_t)br0 * N + n0 + bc0);
  breg[1] = *(const u32x4*)(B + (size_t)br1 * N + n0 + bc1);
  {
    const __bf16* e0 = (const __bf16*)&breg[0];
    const __bf16* e1 = (const __bf16*)&breg[1];
    #pragma unroll
    for (int j = 0; j < 8; ++j) {
      sB[0][(size_t)(bc0 + j) * G_LDB + br0] = e0[j];
      sB[0][(size_t)(bc1 + j) * G_LDB + br1] = e1[j];
    }
  }
  __builtin_amdgcn_s_wait_tensorcnt(0);
  __syncthreads();

  for (int i = 0; i < T; ++i) {
    const int cur = i & 1;
    const bool more = (i + 1) < T;
    if (more) {
      const int k1 = (i + 1) * 32;
      if (wid == 0)
        tdm_load_2d_bf16(A + (size_t)m0 * K + k1, sA_off[cur ^ 1],
                         32, 128, (unsigned)K, (unsigned)M, (unsigned)K,
                         1, 3, 3);
      breg[0] = *(const u32x4*)(B + (size_t)(k1 + br0) * N + n0 + bc0);
      breg[1] = *(const u32x4*)(B + (size_t)(k1 + br1) * N + n0 + bc1);
    }

    // ---- WMMA burst on tile i ----
    Frag bfr[2];
    #pragma unroll
    for (int nt = 0; nt < 2; ++nt)
      load_frag_b(bfr[nt], sB[cur], wn * 32 + nt * 16 + ln, hlf, G_LDB);
    #pragma unroll
    for (int mt = 0; mt < 4; ++mt) {
      Frag af;
      load_frag_a(af, sA[cur], wm * 64 + mt * 16 + ln, hlf, G_LDA);
      #pragma unroll
      for (int nt = 0; nt < 2; ++nt)
        acc[mt][nt] = wmma_bf16(af, bfr[nt], acc[mt][nt]);
    }

    if (more) {
      const __bf16* e0 = (const __bf16*)&breg[0];
      const __bf16* e1 = (const __bf16*)&breg[1];
      #pragma unroll
      for (int j = 0; j < 8; ++j) {
        sB[cur ^ 1][(size_t)(bc0 + j) * G_LDB + br0] = e0[j];
        sB[cur ^ 1][(size_t)(bc1 + j) * G_LDB + br1] = e1[j];
      }
      __builtin_amdgcn_s_wait_tensorcnt(0);
      __syncthreads();
    }
  }

  // Epilogue. C layout: VGPR r -> row = r + 8*half, col = lane%16.
  #pragma unroll
  for (int mt = 0; mt < 4; ++mt) {
    #pragma unroll
    for (int nt = 0; nt < 2; ++nt) {
      const int col = n0 + wn * 32 + nt * 16 + ln;
      float bv = 0.f;
      if constexpr (BIAS) bv = bias[col];
      #pragma unroll
      for (int r = 0; r < 8; ++r) {
        const int row = m0 + wm * 64 + mt * 16 + r + 8 * hlf;
        const size_t idx = (size_t)row * N + col;
        float v = acc[mt][nt][r] * oscale + bv;
        if constexpr (RES) v += res[idx];
        if constexpr (GELU) v = 0.5f * v * (1.f + erff(v * 0.70710678118654752f));
        if constexpr (OUTF) outF[idx] = v;
        else                outB[idx] = (__bf16)v;
      }
    }
  }
}

// ---------------------------------------------------------------------------
// Flash attention. H=16, hd=64, N1=N2=1024. q is PRE-SCALED by 1/sqrt(hd).
//   q : [B*1024, 1024] bf16, head h at cols h*64..h*64+63
//   kv: [B*1024, 2048] bf16, K at cols h*64.., V at cols 1024+h*64..
//   o : [B*1024, 1024] bf16
// grid = (16 q-blocks of 64 rows, B*H). block = 128 thr (4 waves x 16 q-rows)
// K tile streamed by TDM; V tile transposed manually; softmax row reductions
// via DPP16 (no LDS traffic).
// ---------------------------------------------------------------------------
__global__ __launch_bounds__(128)
void attn_kernel(const __bf16* __restrict__ q, const __bf16* __restrict__ kv,
                 __bf16* __restrict__ o) {
  const int NSEQ = 1024;
  const int bh = blockIdx.y;
  const int b = bh >> 4, h = bh & 15;
  const int tid = threadIdx.x, wid = tid >> 5, lane = tid & 31;
  const int hlf = lane >> 4, ln = lane & 15;
  const int qrow = blockIdx.x * 64 + wid * 16;

  const __bf16* qb = q  + (size_t)(b * NSEQ) * 1024 + h * 64;
  const __bf16* kb = kv + (size_t)(b * NSEQ) * 2048 + h * 64;
  const __bf16* vb = kb + 1024;

  __shared__ __attribute__((aligned(16))) __bf16 sK[32 * 64];       // [kv][d]
  __shared__ __attribute__((aligned(16))) __bf16 sV[64 * 40];       // [d][kv]
  __shared__ __attribute__((aligned(16))) __bf16 sP[4][16 * 40];    // P per wave
  const unsigned sK_off = lds_offset(sK);

  // Persistent Q fragments (16 rows x 64 K-dim = 2 A-frags) from HBM
  Frag qf[2];
  #pragma unroll
  for (int kblk = 0; kblk < 2; ++kblk) {
    const __bf16* p = qb + (size_t)(qrow + ln) * 1024 + kblk * 32 + 8 * hlf;
    qf[kblk].q[0] = *(const u32x4*)(p);
    qf[kblk].q[1] = *(const u32x4*)(p + 16);
  }

  float m_s[8], l_s[8];
  #pragma unroll
  for (int r = 0; r < 8; ++r) { m_s[r] = -1e30f; l_s[r] = 0.f; }
  v8f oacc[4] = {};

  for (int kv0 = 0; kv0 < NSEQ; kv0 += 32) {
    __syncthreads();
    // K tile 32x64 via TDM (row-major, no padding)
    if (wid == 0)
      tdm_load_2d_bf16(kb + (size_t)kv0 * 2048, sK_off,
                       64, 32, 2048u, (unsigned)NSEQ, 2048u, 0, 0, 0);
    // V tile transposed into sV[d][kv]
    {
      int r = tid >> 3, c = (tid & 7) * 8;    // r:0..15, c:0..56
      #pragma unroll
      for (int rr = 0; rr < 2; ++rr) {
        int kr = r + rr * 16;
        u32x4 d4 = *(const u32x4*)(vb + (size_t)(kv0 + kr) * 2048 + c);
        const __bf16* e = (const __bf16*)&d4;
        #pragma unroll
        for (int j = 0; j < 8; ++j) sV[(size_t)(c + j) * 40 + kr] = e[j];
      }
    }
    __builtin_amdgcn_s_wait_tensorcnt(0);
    __syncthreads();

    // S = Q @ K^T : two 16x16 tiles (kv cols 0-15, 16-31), K-dim = 64
    v8f s[2] = {};
    #pragma unroll
    for (int nt = 0; nt < 2; ++nt) {
      #pragma unroll
      for (int db = 0; db < 2; ++db) {
        Frag kf;
        const __bf16* p = sK + (size_t)(nt * 16 + ln) * 64 + db * 32 + 8 * hlf;
        kf.q[0] = *(const u32x4*)(p);
        kf.q[1] = *(const u32x4*)(p + 16);
        s[nt] = wmma_bf16(qf[db], kf, s[nt]);
      }
    }

    // Online softmax (row = r + 8*half; 16 lanes of a half hold 16 cols)
    float sf[8];
    #pragma unroll
    for (int r = 0; r < 8; ++r) {
      float s0 = s[0][r], s1 = s[1][r];
      float rm = row16_max(fmaxf(s0, s1));
      float mn = fmaxf(m_s[r], rm);
      float p0 = __expf(s0 - mn), p1 = __expf(s1 - mn);
      float rs = row16_sum(p0 + p1);
      float f = __expf(m_s[r] - mn);
      l_s[r] = l_s[r] * f + rs;
      m_s[r] = mn;
      sf[r] = f;
      const int row = r + 8 * hlf;
      sP[wid][row * 40 + ln]      = (__bf16)p0;
      sP[wid][row * 40 + 16 + ln] = (__bf16)p1;
    }
    #pragma unroll
    for (int dt = 0; dt < 4; ++dt)
      #pragma unroll
      for (int r = 0; r < 8; ++r) oacc[dt][r] *= sf[r];

    // O += P(16x32) @ V(32x64)
    Frag pf;
    load_frag_a(pf, &sP[wid][0], ln, hlf, 40);
    #pragma unroll
    for (int dt = 0; dt < 4; ++dt) {
      Frag vf;
      load_frag_b(vf, sV, dt * 16 + ln, hlf, 40);
      oacc[dt] = wmma_bf16(pf, vf, oacc[dt]);
    }
  }

  #pragma unroll
  for (int r = 0; r < 8; ++r) l_s[r] = 1.f / l_s[r];
  __bf16* ob = o + (size_t)(b * NSEQ) * 1024 + h * 64;
  #pragma unroll
  for (int dt = 0; dt < 4; ++dt)
    #pragma unroll
    for (int r = 0; r < 8; ++r) {
      const int row = qrow + r + 8 * hlf;
      ob[(size_t)row * 1024 + dt * 16 + ln] = (__bf16)(oacc[dt][r] * l_s[r]);
    }
}

// ---------------------------------------------------------------------------
// Host orchestration
// ---------------------------------------------------------------------------
extern "C" void kernel_launch(void* const* d_in, const int* in_sizes, int n_in,
                              void* d_out, int out_size, void* d_ws, size_t ws_size,
                              hipStream_t stream) {
  (void)in_sizes; (void)n_in; (void)out_size; (void)ws_size;
  const float* x      = (const float*)d_in[0];
  const float* y      = (const float*)d_in[1];
  const float* ln1_w  = (const float*)d_in[2];
  const float* ln1_b  = (const float*)d_in[3];
  const float* ln2_w  = (const float*)d_in[4];
  const float* ln2_b  = (const float*)d_in[5];
  const float* ln3_w  = (const float*)d_in[6];
  const float* ln3_b  = (const float*)d_in[7];
  const float* Wq     = (const float*)d_in[8];
  const float* Wkv    = (const float*)d_in[9];
  const float* Wproj  = (const float*)d_in[10];
  const float* bproj  = (const float*)d_in[11];
  const float* W1     = (const float*)d_in[12];
  const float* b1     = (const float*)d_in[13];
  const float* W2     = (const float*)d_in[14];
  const float* b2     = (const float*)d_in[15];
  float* out = (float*)d_out;

  const int M = 4096;            // B * N1
  char* ws = (char*)d_ws;
  size_t off = 0;
  auto alloc = [&](size_t bytes) -> void* {
    void* p = ws + off;
    off += (bytes + 255) & ~(size_t)255;
    return p;
  };
  __bf16* wq_bf    = (__bf16*)alloc((size_t)1024 * 1024 * 2);
  __bf16* wkv_bf   = (__bf16*)alloc((size_t)768 * 2048 * 2);
  __bf16* wproj_bf = (__bf16*)alloc((size_t)1024 * 1024 * 2);
  __bf16* w1_bf    = (__bf16*)alloc((size_t)1024 * 4096 * 2);
  __bf16* w2_bf    = (__bf16*)alloc((size_t)4096 * 1024 * 2);
  __bf16* xn       = (__bf16*)alloc((size_t)M * 1024 * 2);
  __bf16* yn       = (__bf16*)alloc((size_t)M * 768 * 2);
  __bf16* qb       = (__bf16*)alloc((size_t)M * 1024 * 2);
  __bf16* kvb      = (__bf16*)alloc((size_t)M * 2048 * 2);
  __bf16* ob       = (__bf16*)alloc((size_t)M * 1024 * 2);
  float*  x1       = (float*) alloc((size_t)M * 1024 * 4);
  __bf16* x1n      = (__bf16*)alloc((size_t)M * 1024 * 2);
  __bf16* hb       = (__bf16*)alloc((size_t)M * 4096 * 2);

  auto cvt = [&](const float* src, __bf16* dst, int n) {
    cvt_bf16_kernel<<<(n + 255) / 256, 256, 0, stream>>>(src, dst, n);
  };
  cvt(Wq,    wq_bf,    1024 * 1024);
  cvt(Wkv,   wkv_bf,   768 * 2048);
  cvt(Wproj, wproj_bf, 1024 * 1024);
  cvt(W1,    w1_bf,    1024 * 4096);
  cvt(W2,    w2_bf,    4096 * 1024);

  // LN1 / LN2
  ln_kernel<<<M, 256, 0, stream>>>(x, ln1_w, ln1_b, xn, 1024);
  ln_kernel<<<M, 256, 0, stream>>>(y, ln2_w, ln2_b, yn, 768);

  // q = (xn @ Wq) * hd^-0.5  (scale folded here; exact in bf16)
  gemm_bf16_kernel<false, false, false, false>
      <<<dim3(M / 128, 1024 / 128), 256, 0, stream>>>(
          xn, wq_bf, M, 1024, 1024, 0.125f, nullptr, nullptr, nullptr, qb);
  // kv = yn @ Wkv
  gemm_bf16_kernel<false, false, false, false>
      <<<dim3(M / 128, 2048 / 128), 256, 0, stream>>>(
          yn, wkv_bf, M, 2048, 768, 1.0f, nullptr, nullptr, nullptr, kvb);

  // o = softmax(q k^T) @ v
  attn_kernel<<<dim3(16, 64), 128, 0, stream>>>(qb, kvb, ob);

  // x1 = x + o @ Wproj + bproj   (f32)
  gemm_bf16_kernel<true, true, false, true>
      <<<dim3(M / 128, 1024 / 128), 256, 0, stream>>>(
          ob, wproj_bf, M, 1024, 1024, 1.0f, bproj, x, x1, nullptr);

  // x1n = LN3(x1)
  ln_kernel<<<M, 256, 0, stream>>>(x1, ln3_w, ln3_b, x1n, 1024);

  // h = gelu(x1n @ W1 + b1)   (bf16)
  gemm_bf16_kernel<true, false, true, false>
      <<<dim3(M / 128, 4096 / 128), 256, 0, stream>>>(
          x1n, w1_bf, M, 4096, 1024, 1.0f, b1, nullptr, nullptr, hb);

  // out = x1 + h @ W2 + b2   (f32)
  gemm_bf16_kernel<true, true, false, true>
      <<<dim3(M / 128, 1024 / 128), 256, 0, stream>>>(
          hb, w2_bf, M, 1024, 4096, 1.0f, b2, x1, out, nullptr);
}
